// ConvKAN_22204980920977
// MI455X (gfx1250) — compile-verified, matrix operand order checked by reading, not run on hardware
//
#include <hip/hip_runtime.h>
#include <hip/hip_bf16.h>

// ---------------------------------------------------------------------------
// ConvKAN fused implicit-GEMM for gfx1250 (wave32, WMMA f16 16x16x32, f32 acc)
//
// out(32768x64) = expand(patches)(32768x5184) @ Wc(64x5184)^T
//   expanded K layout: k = f*9 + j ; j==0 -> silu, j=1..8 -> cubic B-spline j-1
// ---------------------------------------------------------------------------

typedef __attribute__((ext_vector_type(16))) _Float16 v16h;
typedef __attribute__((ext_vector_type(8)))  float    v8f;
typedef __attribute__((ext_vector_type(8)))  int      v8i;

#define B_DIM   8
#define C_IN    64
#define H_DIM   64
#define W_DIM   64
#define OUTC    64
#define INFEAT  576            // C_IN*3*3
#define KEXP    9              // 1 silu + 8 spline values per raw feature
#define KTOT    (INFEAT*KEXP)  // 5184
#define KSTEPS  (KTOT/32)      // 162

#define MT      64             // pixels per block (4 waves x 16 rows)
#define CF      32             // raw features per chunk
#define KC      (CF*KEXP)      // 288 expanded-K per chunk (9 WMMA K-steps)
#define NCHUNK  (INFEAT/CF)    // 18
#define ASTRIDE (KC+8)         // 296 halves: padded LDS row stride (dword-even)

// ---------------------------------------------------------------------------
// Prep: build Wc in f16, stored directly in WMMA B-fragment order:
//   half index = ((S*4 + t)*32 + lane)*16 + e
//   lane<16: N = t*16+lane,    K = S*32 + (e<8 ? e : 8+e)   pattern below
//   lane>=16: same N = t*16+(lane-16), K offset +8 / +24 halves
// Matches ISA 16-bit A/B layout: vgpr v=e/2, half=e&1,
//   K = S*32 + (lane<16?0:8) + (v<4 ? 2v+half : 16+2(v-4)+half)
// ---------------------------------------------------------------------------
__global__ __launch_bounds__(256) void convkan_prep_w(
    const float* __restrict__ bw,   // (64,576)
    const float* __restrict__ sw,   // (64,576,8)
    const float* __restrict__ sc,   // (64,576)
    _Float16* __restrict__ wq)
{
    int id = blockIdx.x * 256 + threadIdx.x;      // 162*4*32*16 = 331776 total
    if (id >= KSTEPS * 4 * 32 * 16) return;
    int e  = id & 15;
    int L  = (id >> 4) & 31;
    int t  = (id >> 9) & 3;
    int S  = id >> 11;
    int v  = e >> 1, hi = e & 1;
    int k  = S * 32 + ((L < 16) ? 0 : 8)
           + ((v < 4) ? (v * 2 + hi) : (16 + (v - 4) * 2 + hi));
    int n  = t * 16 + (L & 15);
    int f  = k / KEXP;
    int j  = k - f * KEXP;
    float val = (j == 0) ? bw[n * INFEAT + f]
                         : sw[(n * INFEAT + f) * 8 + (j - 1)] * sc[n * INFEAT + f];
    wq[id] = (_Float16)val;
}

// ---------------------------------------------------------------------------
// Closed-form uniform cubic B-spline: grid knots t_i = -2.2 + 0.4*i, i=0..11.
// s = (x+2.2)*2.5 ; cell c = floor(s) in [0,10] ; 4 nonzero bases c-3..c.
// ---------------------------------------------------------------------------
__device__ __forceinline__ void spline8(float x, float bs[8]) {
    float s  = (x + 2.2f) * 2.5f;
    int   c  = (int)floorf(s);
    bool  ok = (s >= 0.f) && (s < 11.f);
    float t  = s - (float)c;
    float omt = 1.f - t;
    float t2 = t * t, t3 = t2 * t;
    float w0 = omt * omt * omt * (1.f / 6.f);
    float w1 = (3.f * t3 - 6.f * t2 + 4.f) * (1.f / 6.f);
    float w2 = (-3.f * t3 + 3.f * t2 + 3.f * t + 1.f) * (1.f / 6.f);
    float w3 = t3 * (1.f / 6.f);
#pragma unroll
    for (int j = 0; j < 8; ++j) {
        int d = c - j;
        float v = (d == 0) ? w3 : (d == 1) ? w2 : (d == 2) ? w1 : (d == 3) ? w0 : 0.f;
        bs[j] = ok ? v : 0.f;
    }
}

// ---------------------------------------------------------------------------
// Main fused kernel: 512 blocks x 128 threads (4 waves). Block = 64 pixels.
// ---------------------------------------------------------------------------
__global__ __launch_bounds__(128) void convkan_main(
    const float*    __restrict__ x,     // (8,64,64,64)
    const _Float16* __restrict__ wq,    // swizzled Wc fragments
    float*          __restrict__ out)   // (8,64,64,64)
{
    __shared__ _Float16 Atile[MT * ASTRIDE];   // 37,888 bytes

    const int tid    = threadIdx.x;
    const int lane   = tid & 31;
    const int wv     = tid >> 5;               // wave 0..3
    const int blockM = blockIdx.x * MT;

    v8f acc0 = {}, acc1 = {}, acc2 = {}, acc3 = {};

    const int hiHalf = (lane & 16) ? 8 : 0;    // K-offset for lanes 16..31
    const int mRow   = wv * 16 + (lane & 15);  // A-fragment row for this lane

    for (int c = 0; c < NCHUNK; ++c) {
        // ---- cooperative expansion: 64 pixels x 32 features -> LDS (f16) ----
        const int f0 = c * CF;
        for (int it = 0; it < (MT * CF) / 128; ++it) {
            int idx = it * 128 + tid;          // 0..2047
            int m   = idx & (MT - 1);          // pixel in tile (coalesced x loads)
            int fl  = idx >> 6;                // 0..31 feature-in-chunk
            int f   = f0 + fl;
            int gm  = blockM + m;
            int b   = gm >> 12, hw = gm & 4095, oh = hw >> 6, ow = hw & 63;
            int ch  = f / 9, r = f - ch * 9, ki = r / 3, kj = r - ki * 3;
            int ih  = oh + ki - 1, iw = ow + kj - 1;
            float v = 0.f;
            if (ih >= 0 && ih < H_DIM && iw >= 0 && iw < W_DIM)
                v = x[(((b << 6) + ch) << 12) + (ih << 6) + iw];
            float si = v * (1.f / (1.f + __expf(-v)));   // silu
            float bs[8];
            spline8(v, bs);
            _Float16* dst = Atile + m * ASTRIDE + fl * KEXP;
            dst[0] = (_Float16)si;
#pragma unroll
            for (int j = 0; j < 8; ++j) dst[1 + j] = (_Float16)bs[j];
        }
        __syncthreads();

        // ---- WMMA: 9 K-steps x 4 N-tiles per wave ----
        const unsigned int* Ar =
            (const unsigned int*)(Atile + mRow * ASTRIDE);
        for (int s9 = 0; s9 < 9; ++s9) {
            int kb = s9 * 32 + hiHalf;         // even
            int d0 = kb >> 1;
            union { v8i i; v16h h; } au;
            au.i[0] = Ar[d0 + 0];  au.i[1] = Ar[d0 + 1];
            au.i[2] = Ar[d0 + 2];  au.i[3] = Ar[d0 + 3];
            au.i[4] = Ar[d0 + 8];  au.i[5] = Ar[d0 + 9];
            au.i[6] = Ar[d0 + 10]; au.i[7] = Ar[d0 + 11];

            int S = c * 9 + s9;
            const _Float16* wp = wq + ((S * 4) * 32 + lane) * 16;  // 32B aligned
            v16h b0 = *(const v16h*)(wp);
            v16h b1 = *(const v16h*)(wp + 512);
            v16h b2 = *(const v16h*)(wp + 1024);
            v16h b3 = *(const v16h*)(wp + 1536);

            acc0 = __builtin_amdgcn_wmma_f32_16x16x32_f16(
                       false, au.h, false, b0, (short)0, acc0, false, false);
            acc1 = __builtin_amdgcn_wmma_f32_16x16x32_f16(
                       false, au.h, false, b1, (short)0, acc1, false, false);
            acc2 = __builtin_amdgcn_wmma_f32_16x16x32_f16(
                       false, au.h, false, b2, (short)0, acc2, false, false);
            acc3 = __builtin_amdgcn_wmma_f32_16x16x32_f16(
                       false, au.h, false, b3, (short)0, acc3, false, false);
        }
        __syncthreads();
    }

    // ---- epilogue: C-layout lane r -> M = r + 8*(lane>=16), N = lane&15 ----
    int mbase = blockM + wv * 16 + ((lane & 16) ? 8 : 0);
    int n     = lane & 15;
    v8f accs[4] = { acc0, acc1, acc2, acc3 };
#pragma unroll
    for (int t = 0; t < 4; ++t) {
        int o = t * 16 + n;
#pragma unroll
        for (int r = 0; r < 8; ++r) {
            int gm = mbase + r;
            int b  = gm >> 12;
            out[(b << 18) + (o << 12) + (gm & 4095)] = accs[t][r];
        }
    }
}

// ---------------------------------------------------------------------------
extern "C" void kernel_launch(void* const* d_in, const int* in_sizes, int n_in,
                              void* d_out, int out_size, void* d_ws, size_t ws_size,
                              hipStream_t stream) {
    const float* x  = (const float*)d_in[0];   // (8,64,64,64)
    const float* bw = (const float*)d_in[1];   // (64,576)
    const float* sw = (const float*)d_in[2];   // (64,576,8)
    const float* sc = (const float*)d_in[3];   // (64,576)
    float* out      = (float*)d_out;
    _Float16* wq    = (_Float16*)d_ws;         // needs 663,552 bytes

    (void)in_sizes; (void)n_in; (void)out_size; (void)ws_size;

    // Build swizzled f16 combined weight matrix (162*4*32*16 halves)
    convkan_prep_w<<<(KSTEPS * 4 * 32 * 16) / 256, 256, 0, stream>>>(bw, sw, sc, wq);

    // Fused expand + GEMM: 32768 pixels / 64 per block
    convkan_main<<<(B_DIM * H_DIM * W_DIM) / MT, 128, 0, stream>>>(x, wq, out);
}
